// QwenAttention_28484223107527
// MI455X (gfx1250) — matrix operand order repaired and checked by reading, not compiled, not run on hardware
//
#include <hip/hip_runtime.h>
#include <hip/hip_bf16.h>
#include <stdint.h>

#define B_   32
#define H_   16
#define D_   128
#define HID_ 2048
#define BS_  16
#define BPS_ 128          // blocks per seq
#define SMAX_ 2048        // BPS_*BS_

typedef __bf16 v16bf __attribute__((ext_vector_type(16)));
typedef float  v8f   __attribute__((ext_vector_type(8)));

// ---------------------------------------------------------------------------
// Fused skinny GEMM  out_z[M x HID] = x[M x HID] @ W_z[HID x HID] + bias_z
// (M = 32).  blockIdx.z selects which of up to 3 weight/bias/out triples.
// One wave (32 threads) computes one 16x16 output tile with 64 bf16 WMMAs.
// A layout (16-bit A 16x32, contiguous k, same as B): lane l<16 holds row m0+l,
//   elems j = K {k0 + j}, j = 0..15;
//   lane l+16 holds K {k0 + 16 + j}.
// B layout (16-bit B 32x16): lane holds column n0+(l&15),
//   elems j = K {k0 + (l<16?0:16) + j}.  W[k][n0+l] loads are coalesced.
// D layout: VGPR r -> M = m0 + r + (l<16?0:8), N = n0 + (l&15).
// ---------------------------------------------------------------------------
__global__ void proj_wmma(const float* __restrict__ x,
                          const float* __restrict__ W0, const float* __restrict__ c0, float* __restrict__ o0,
                          const float* __restrict__ W1, const float* __restrict__ c1, float* __restrict__ o1,
                          const float* __restrict__ W2, const float* __restrict__ c2, float* __restrict__ o2)
{
    const float* W;  const float* bias;  float* out;
    if      (blockIdx.z == 0) { W = W0; bias = c0; out = o0; }
    else if (blockIdx.z == 1) { W = W1; bias = c1; out = o1; }
    else                      { W = W2; bias = c2; out = o2; }

    const int n0   = blockIdx.x * 16;
    const int m0   = blockIdx.y * 16;
    const int lane = threadIdx.x & 31;
    const int half = lane >> 4;      // 0 | 1
    const int l    = lane & 15;

    v8f acc = {};
    const int rowA = m0 + l;

    for (int k0 = 0; k0 < HID_; k0 += 32) {
        v16bf a, b;

        // ---- A tile: x[rowA][kA + j] -------------------------------------
        const int    kA = k0 + half * 16;
        const float* xa = x + (size_t)rowA * HID_ + kA;
#pragma unroll
        for (int j = 0; j < 8; ++j) {
            a[j]     = (__bf16)xa[j];
            a[8 + j] = (__bf16)xa[8 + j];
        }

        // ---- B tile: W[kB + j][n0 + l]  (coalesced across lanes) ---------
        const int    kB = k0 + half * 16;
        const float* wb = W + (size_t)kB * HID_ + n0 + l;
#pragma unroll
        for (int j = 0; j < 16; ++j)
            b[j] = (__bf16)wb[(size_t)j * HID_];

        // prefetch next weight K-slab into cache (global_prefetch_b8)
        __builtin_prefetch((const void*)(wb + (size_t)32 * HID_), 0, 1);

        acc = __builtin_amdgcn_wmma_f32_16x16x32_bf16(
                  false, a, false, b, (short)0, acc, false, false);
    }

    // ---- write D + bias --------------------------------------------------
#pragma unroll
    for (int r = 0; r < 8; ++r) {
        const int m = m0 + r + half * 8;
        const int n = n0 + l;
        out[(size_t)m * HID_ + n] = acc[r] + bias[n];
    }
}

// ---------------------------------------------------------------------------
// Paged decode attention for one (b, h): scores -> softmax -> PV.
// Fresh k/v (position seq_len-1) substituted from workspace so the input
// caches are never mutated.  q row staged into LDS by wave 0 with a single
// global_load_async_to_lds_b128 (32 lanes x 16B = 512B).
// ---------------------------------------------------------------------------
__global__ void attn_kernel(const float* __restrict__ qb,
                            const float* __restrict__ kb,
                            const float* __restrict__ vb,
                            const float* __restrict__ kc,
                            const float* __restrict__ vc,
                            const int*   __restrict__ block_table,
                            const int*   __restrict__ seq_lens,
                            float*       __restrict__ outb)
{
    const int b = blockIdx.x;
    const int h = blockIdx.y;
    const int t = threadIdx.x;            // 0..255

    __shared__ float qs[D_];
    __shared__ float sc[SMAX_];
    __shared__ float red[256];

    const int   seq = seq_lens[b];
    const int   pos = seq - 1;
    const float scale = 0.08838834764831845f;   // 1/sqrt(128)

    // ---- async-stage q row (512B) into LDS: wave 0 only ------------------
    if (t < 32) {
        const uint64_t gsrc = (uint64_t)(uintptr_t)(qb + ((size_t)b * H_ + h) * D_) +
                              (uint64_t)t * 16u;
        const uint32_t ldst = (uint32_t)(uintptr_t)(&qs[0]) + (uint32_t)t * 16u;
        asm volatile("global_load_async_to_lds_b128 %0, %1, off"
                     :: "v"(ldst), "v"(gsrc) : "memory");
        asm volatile("s_wait_asynccnt 0x0" ::: "memory");
    }
    __syncthreads();

    // ---- pass 1: scores --------------------------------------------------
    float lmax = -1e30f;
    for (int s = t; s < SMAX_; s += 256) {
        float sco = -1e30f;
        if (s < seq) {
            const float* krow;
            if (s == pos) {
                krow = kb + ((size_t)b * H_ + h) * D_;
            } else {
                const int blk = block_table[b * BPS_ + (s >> 4)];
                krow = kc + (((size_t)blk * BS_ + (s & 15)) * H_ + h) * D_;
            }
            float accd = 0.f;
#pragma unroll 8
            for (int d = 0; d < D_; d += 4) {
                const float4 kv = *(const float4*)(krow + d);
                accd += qs[d] * kv.x + qs[d + 1] * kv.y +
                        qs[d + 2] * kv.z + qs[d + 3] * kv.w;
            }
            sco = accd * scale;
        }
        sc[s] = sco;
        lmax  = fmaxf(lmax, sco);
    }
    red[t] = lmax;
    __syncthreads();
#pragma unroll
    for (int off = 128; off > 0; off >>= 1) {
        if (t < off) red[t] = fmaxf(red[t], red[t + off]);
        __syncthreads();
    }
    const float gmax = red[0];
    __syncthreads();

    // ---- pass 2: exp + sum ----------------------------------------------
    float lsum = 0.f;
    for (int s = t; s < SMAX_; s += 256) {
        const float p = (s < seq) ? __expf(sc[s] - gmax) : 0.f;
        sc[s] = p;
        lsum += p;
    }
    red[t] = lsum;
    __syncthreads();
#pragma unroll
    for (int off = 128; off > 0; off >>= 1) {
        if (t < off) red[t] += red[t + off];
        __syncthreads();
    }
    const float inv = 1.0f / red[0];
    __syncthreads();

    // ---- pass 3: PV (dim-coalesced) -------------------------------------
    const int d   = t & (D_ - 1);
    const int grp = t >> 7;               // 0 | 1
    float acc = 0.f;
    for (int s = grp; s < seq; s += 2) {
        const float* vrow;
        if (s == pos) {
            vrow = vb + ((size_t)b * H_ + h) * D_;
        } else {
            const int blk = block_table[b * BPS_ + (s >> 4)];
            vrow = vc + (((size_t)blk * BS_ + (s & 15)) * H_ + h) * D_;
        }
        acc += sc[s] * vrow[d];
    }
    red[t] = acc;
    __syncthreads();
    if (t < D_)
        outb[((size_t)b * H_ + h) * D_ + d] = (red[t] + red[t + 128]) * inv;
}

// ---------------------------------------------------------------------------
extern "C" void kernel_launch(void* const* d_in, const int* in_sizes, int n_in,
                              void* d_out, int out_size, void* d_ws, size_t ws_size,
                              hipStream_t stream)
{
    const float* x   = (const float*)d_in[0];
    const float* Wq  = (const float*)d_in[1];
    const float* bq  = (const float*)d_in[2];
    const float* Wk  = (const float*)d_in[3];
    const float* bk  = (const float*)d_in[4];
    const float* Wv  = (const float*)d_in[5];
    const float* bv  = (const float*)d_in[6];
    const float* Wo  = (const float*)d_in[7];
    const float* bo  = (const float*)d_in[8];
    const float* kc  = (const float*)d_in[9];
    const float* vc  = (const float*)d_in[10];
    // d_in[11] position_ids, d_in[12] slot_mapping: redundant (derivable)
    const int* block_table = (const int*)d_in[13];
    const int* seq_lens    = (const int*)d_in[14];

    float* qbuf = (float*)d_ws;
    float* kbuf = qbuf + (size_t)B_ * HID_;
    float* vbuf = kbuf + (size_t)B_ * HID_;
    float* abuf = vbuf + (size_t)B_ * HID_;
    float* outp = (float*)d_out;

    // Fused QKV: 128 n-tiles x 2 m-tiles x 3 matrices = 768 waves in flight.
    proj_wmma<<<dim3(HID_ / 16, B_ / 16, 3), 32, 0, stream>>>(
        x, Wq, bq, qbuf, Wk, bk, kbuf, Wv, bv, vbuf);

    attn_kernel<<<dim3(B_, H_), 256, 0, stream>>>(qbuf, kbuf, vbuf, kc, vc,
                                                  block_table, seq_lens, abuf);

    // Output projection (single matrix; z-dim = 1).
    proj_wmma<<<dim3(HID_ / 16, B_ / 16, 1), 32, 0, stream>>>(
        abuf, Wo, bo, outp, Wo, bo, outp, Wo, bo, outp);
}
